// MultiRelationalGCN_13898514170200
// MI455X (gfx1250) — matrix operand order repaired and checked by reading, not compiled, not run on hardware
//
#include <hip/hip_runtime.h>
#include <stdint.h>

typedef _Float16 half4v __attribute__((ext_vector_type(4)));
typedef _Float16 half8  __attribute__((ext_vector_type(8)));
typedef _Float16 half16 __attribute__((ext_vector_type(16)));
typedef float    f32x4  __attribute__((ext_vector_type(4)));
typedef float    f32x8  __attribute__((ext_vector_type(8)));
typedef uint32_t u32x4  __attribute__((ext_vector_type(4)));
typedef uint32_t u32x8  __attribute__((ext_vector_type(8)));

namespace {
constexpr int R_REL = 2;
constexpr int T_DIM = 128;
constexpr int N_DIM = 128;
constexpr int C_IN  = 64;
constexpr int C_OUT = 64;
constexpr int BDIM  = 256;   // 8 wave32

// LDS layout (f16 strides padded +8 halfs: row stride 272B = 68 dwords so the
// 16 A-fragment rows land on disjoint bank groups)
constexpr int XH_S = 136;                          // 64  rows (staged x, f16)
constexpr int YH_S = 136;                          // 128 rows (conv result, f16)
constexpr int AH_S = 136;                          // 128 rows (normalized A, f16)
constexpr int OFF_XH   = 0;
constexpr int OFF_YH   = OFF_XH + 64  * XH_S * 2;  // 17408
constexpr int OFF_AH   = OFF_YH + 128 * YH_S * 2;  // 52224
constexpr int OFF_ARAW = OFF_AH + 128 * AH_S * 2;  // 87040: raw A slice (f32, TDM dst)
constexpr int OFF_DV   = OFF_ARAW + 128 * 128 * 4; // 152576 (128 f32 rsqrt-degrees)
constexpr int OFF_BB   = OFF_DV + 128 * 4;         // 153088 (128 f32 bias)
constexpr int LDS_BYTES = OFF_BB + 128 * 4;        // 153600 B -> 2 WGs / WGP
}

__device__ inline half16 make_h16(half8 lo, half8 hi) {
  half16 r;
#pragma unroll
  for (int i = 0; i < 8; ++i) { r[i] = lo[i]; r[i + 8] = hi[i]; }
  return r;
}

// A-operand (16x32 f16): lane holds M=lane%16, K = (lane/16)*8 + {0..7,16..23}
__device__ inline half16 lds_afrag(const _Float16* p) {
  return make_h16(*(const half8*)p, *(const half8*)(p + 16));
}
// B-operand (32x16 f16): lane holds N=lane%16, K = (lane/16)*8 + {0..7,16..23}
__device__ inline half16 lds_bfrag(const _Float16* p, int ld) {
  half16 r;
#pragma unroll
  for (int i = 0; i < 8; ++i) { r[i] = p[i * ld]; r[i + 8] = p[(16 + i) * ld]; }
  return r;
}

__device__ inline f32x8 wmma_f16(half16 a, half16 b, f32x8 c) {
  return __builtin_amdgcn_wmma_f32_16x16x32_f16(
      /*neg_a=*/false, a, /*neg_b=*/false, b,
      /*c_mod=*/(short)0, c, /*reuse_a=*/false, /*reuse_b=*/false);
}

// Tensor Data Mover: DMA a 2D f32 tile (tile_d1 rows x tile_d0 elems, row
// stride in elements) from global memory into LDS at lds_off. D# per
// CDNA5 ISA; 2D tensor -> groups 0 and 1 only
// (VADDR2/VADDR3 = NULL). Tracked by TENSORcnt.
__device__ inline void tdm_load_2d_f32(uint32_t lds_off, const void* gptr,
                                       uint32_t tensor_d0, uint32_t tensor_d1,
                                       uint32_t tile_d0, uint32_t tile_d1,
                                       uint32_t row_stride_elems) {
  const uint64_t ga = (uint64_t)gptr;
  u32x4 g0;
  g0[0] = 1u;                                            // count=1, user mode
  g0[1] = lds_off;                                       // lds_addr (bytes)
  g0[2] = (uint32_t)ga;                                  // global_addr[31:0]
  g0[3] = (uint32_t)((ga >> 32) & 0x01FFFFFFu)           // global_addr[56:32]
          | (2u << 30);                                  // type = 2 ("image")
  u32x8 g1;
  g1[0] = 2u << 16;                                      // data_size = 2 (4 bytes)
  g1[1] = (tensor_d0 & 0xFFFFu) << 16;                   // tensor_dim0[15:0]
  g1[2] = ((tensor_d0 >> 16) & 0xFFFFu)                  // tensor_dim0[31:16]
          | ((tensor_d1 & 0xFFFFu) << 16);               // tensor_dim1[15:0]
  g1[3] = ((tensor_d1 >> 16) & 0xFFFFu)                  // tensor_dim1[31:16]
          | (tile_d0 << 16);                             // tile_dim0
  g1[4] = tile_d1;                                       // tile_dim1 (tile_dim2=0)
  g1[5] = row_stride_elems;                              // tensor_dim0_stride[31:0]
  g1[6] = 0u;                                            // d0_stride hi / d1_stride lo
  g1[7] = 0u;                                            // d1_stride hi (unused, 2D)
  asm volatile("tensor_load_to_lds %0, %1" :: "s"(g0), "s"(g1) : "memory");
}

__global__ __launch_bounds__(BDIM) void mrgcn_fused_kernel(
    const float* __restrict__ xg, const float* __restrict__ Ag,
    const float* __restrict__ Wg, const float* __restrict__ bg,
    float* __restrict__ out_x, float* __restrict__ out_A)
{
  extern __shared__ __align__(16) char smem[];
  _Float16* Xh   = (_Float16*)(smem + OFF_XH);
  _Float16* Yh   = (_Float16*)(smem + OFF_YH);
  _Float16* Ah   = (_Float16*)(smem + OFF_AH);
  float*    Araw = (float*)(smem + OFF_ARAW);
  float*    dv   = (float*)(smem + OFF_DV);
  float*    bb   = (float*)(smem + OFF_BB);

  const int tid  = threadIdx.x;
  const int lane = tid & 31;
  const int wave = tid >> 5;
  const int bt   = blockIdx.x;            // one block per (b, t)
  const int b    = bt >> 7;
  const int t    = bt & (T_DIM - 1);

  const size_t abase = (((size_t)b * R_REL) * T_DIM + t) * (size_t)N_DIM * N_DIM;

  // Kick off TDM for the r=0 adjacency slice; it runs async under x-staging
  // and the conv WMMA phase. One wave issues (TDM ignores EXEC; every wave
  // executing it would launch a duplicate DMA).
  if (wave == 0)
    tdm_load_2d_f32(OFF_ARAW, Ag + abase, N_DIM, N_DIM, N_DIM, N_DIM, N_DIM);

  // ---- Phase A: stage x[b,:,t,:] (64x128 f32) into LDS as f16 -------------
  for (int i = tid; i < C_IN * (N_DIM / 4); i += BDIM) {
    const int c  = i >> 5;
    const int n4 = i & 31;
    f32x4 v = *(const f32x4*)(xg + (((size_t)b * C_IN + c) * T_DIM + t) * N_DIM + n4 * 4);
    half4v h;
#pragma unroll
    for (int j = 0; j < 4; ++j) h[j] = (_Float16)v[j];
    *(half4v*)(Xh + c * XH_S + n4 * 4) = h;
  }
  if (tid < R_REL * C_OUT) bb[tid] = bg[tid];
  __syncthreads();

  // ---- Phase B: 1x1 conv  Y(128x128) = W(128x64) @ X(64x128) + b ----------
  {
    const int mr = wave * 16 + (lane & 15);
    const int kb = (lane >> 4) * 8;
    half16 afr[2];
#pragma unroll
    for (int kt = 0; kt < 2; ++kt) {       // K = 64 -> two k-steps of 32
      const float* wp = Wg + mr * C_IN + kt * 32 + kb;
      half16 a;
#pragma unroll
      for (int j = 0; j < 8; ++j) {
        a[j]     = (_Float16)wp[j];
        a[j + 8] = (_Float16)wp[16 + j];
      }
      afr[kt] = a;
    }
    const int obase = wave * 16 + (lane >> 4) * 8;
    const int nc    = lane & 15;
#pragma unroll
    for (int nt = 0; nt < 8; ++nt) {
      f32x8 c;
#pragma unroll
      for (int i = 0; i < 8; ++i) c[i] = 0.0f;
      c = wmma_f16(afr[0], lds_bfrag(Xh + (0  + kb) * XH_S + nt * 16 + nc, XH_S), c);
      c = wmma_f16(afr[1], lds_bfrag(Xh + (32 + kb) * XH_S + nt * 16 + nc, XH_S), c);
#pragma unroll
      for (int i = 0; i < 8; ++i)
        Yh[(obase + i) * YH_S + nt * 16 + nc] = (_Float16)(c[i] + bb[obase + i]);
    }
  }
  if (wave == 0) __builtin_amdgcn_s_wait_tensorcnt(0);  // r=0 slice landed
  __syncthreads();                                      // publish Yh + Araw

  // x_out accumulators: wave w owns D tiles (ct = w/2, nt = (w&1)*4 + 0..3)
  f32x8 acc[4];
#pragma unroll
  for (int j = 0; j < 4; ++j)
#pragma unroll
    for (int i = 0; i < 8; ++i) acc[j][i] = 0.0f;

  for (int r = 0; r < R_REL; ++r) {
    float* Ao = out_A + abase + (size_t)r * N_DIM * N_DIM;

    // pass 1: degrees d[m] = rsqrt(1 + sum_n A[m][n]) from the LDS tile;
    // fused passthrough copy of A to the second output (LDS -> global).
#pragma unroll 1
    for (int i = 0; i < 16; ++i) {
      const int m = wave * 16 + i;
      f32x4 v = *(const f32x4*)(Araw + m * N_DIM + lane * 4);
      *(f32x4*)(Ao + m * N_DIM + lane * 4) = v;
      float s = v[0] + v[1] + v[2] + v[3];
#pragma unroll
      for (int off = 16; off > 0; off >>= 1) s += __shfl_xor(s, off, 32);
      if (lane == 0) dv[m] = rsqrtf(s + 1.0f);
    }
    __syncthreads();

    // pass 2: A_norm = (A + I) * d_m * d_n -> f16 WMMA B-tile in LDS
#pragma unroll 1
    for (int it = 0; it < 16; ++it) {
      const int chunk = tid + it * BDIM;
      const int m  = chunk >> 5;
      const int n4 = chunk & 31;
      f32x4 v = *(const f32x4*)(Araw + m * N_DIM + n4 * 4);
      const float dm = dv[m];
      half4v h;
#pragma unroll
      for (int j = 0; j < 4; ++j) {
        const int n = n4 * 4 + j;
        const float e = v[j] + ((n == m) ? 1.0f : 0.0f);
        h[j] = (_Float16)(e * dm * dv[n]);
      }
      *(half4v*)(Ah + m * AH_S + n4 * 4) = h;
    }
    __syncthreads();   // Araw fully consumed; Ah complete

    // Araw is dead now: start the r=1 DMA so it overlaps the r=0 WMMA phase.
    if (r == 0 && wave == 0)
      tdm_load_2d_f32(OFF_ARAW, Ag + abase + (size_t)N_DIM * N_DIM,
                      N_DIM, N_DIM, N_DIM, N_DIM, N_DIM);

    // pass 3: acc += Y_r(64x128) @ A_norm(128x128), K = 128 -> 4 WMMA k-steps
    {
      const int arow = r * 64 + (wave >> 1) * 16 + (lane & 15);
      const int kb   = (lane >> 4) * 8;
      const int nb   = (wave & 1) * 4;
#pragma unroll
      for (int kt = 0; kt < 4; ++kt) {
        half16 a = lds_afrag(Yh + arow * YH_S + kt * 32 + kb);
#pragma unroll
        for (int j = 0; j < 4; ++j)
          acc[j] = wmma_f16(a, lds_bfrag(Ah + (kt * 32 + kb) * AH_S + (nb + j) * 16 + (lane & 15), AH_S),
                            acc[j]);
      }
    }
    if (r == 0 && wave == 0) __builtin_amdgcn_s_wait_tensorcnt(0);
    __syncthreads();   // Ah/dv reused next relation; r=1 Araw published
  }

  // ---- store x_out (f32). C/D layout: M = i + 8*(lane>>4), N = lane&15 ----
  {
    const int c0 = (wave >> 1) * 16 + (lane >> 4) * 8;
    const int nb = (wave & 1) * 4;
#pragma unroll
    for (int j = 0; j < 4; ++j) {
      const int n = (nb + j) * 16 + (lane & 15);
#pragma unroll
      for (int i = 0; i < 8; ++i)
        out_x[(((size_t)b * C_OUT + (c0 + i)) * T_DIM + t) * N_DIM + n] = acc[j][i];
    }
  }
}

extern "C" void kernel_launch(void* const* d_in, const int* in_sizes, int n_in,
                              void* d_out, int out_size, void* d_ws, size_t ws_size,
                              hipStream_t stream) {
  (void)in_sizes; (void)n_in; (void)out_size; (void)d_ws; (void)ws_size;
  const float* xg = (const float*)d_in[0];
  const float* Ag = (const float*)d_in[1];
  const float* Wg = (const float*)d_in[2];
  const float* bg = (const float*)d_in[3];

  float* out_x = (float*)d_out;                       // (32,64,128,128)
  float* out_A = out_x + (size_t)32 * 64 * 128 * 128; // passthrough (32,2,128,128,128)

  hipFuncSetAttribute(reinterpret_cast<const void*>(mrgcn_fused_kernel),
                      hipFuncAttributeMaxDynamicSharedMemorySize, LDS_BYTES);

  mrgcn_fused_kernel<<<32 * 128, BDIM, LDS_BYTES, stream>>>(
      xg, Ag, Wg, bg, out_x, out_A);
}
